// TransConv_20727512170563
// MI455X (gfx1250) — matrix operand order repaired and checked by reading, not compiled, not run on hardware
//
#include <hip/hip_runtime.h>
#include <hip/hip_bf16.h>
#include <cstdint>

// ---------------------------------------------------------------------------
// TransConv pipeline for MI455X (gfx1250), wave32 + WMMA bf16.
// Async global->LDS staging (ASYNCcnt DMA), double-buffered so the DMA for
// k-step i+1 overlaps the WMMA work of k-step i.
// ---------------------------------------------------------------------------

typedef __bf16 bf16_t;
typedef __attribute__((ext_vector_type(16))) __bf16 v16bf;
typedef __attribute__((ext_vector_type(8)))  __bf16 v8bf;
typedef __attribute__((ext_vector_type(8)))  float  v8f;

#define N_ROWS 100000
#define MPAD   100096   // N_ROWS padded to multiple of 64 (pad rows kept zero)
#define DIM    512
#define BM     128
#define BN     128
#define BK     64
#define PA     72       // LDS pitch (elements) for A tiles (rows x BK)
#define PB     72       // LDS pitch (elements) for B tiles (rows x BK)

#ifndef USE_ASYNC_STAGING
#define USE_ASYNC_STAGING 1
#endif

__device__ __forceinline__ v8bf zero8() {
  v8bf z;
#pragma unroll
  for (int e = 0; e < 8; ++e) z[e] = (bf16_t)0.0f;
  return z;
}

// 16-byte global -> LDS staging copy.  Async path uses the CDNA5 direct
// global->LDS DMA (tracked by ASYNCcnt); fallback is load + ds_store_b128.
__device__ __forceinline__ void stage16(const bf16_t* g, bf16_t* l) {
#if USE_ASYNC_STAGING
  asm volatile("global_load_async_to_lds_b128 %0, %1, off"
               :: "v"((unsigned int)(uintptr_t)l),
                  "v"((unsigned long long)(uintptr_t)g)
               : "memory");
#else
  *(v8bf*)l = *(const v8bf*)g;
#endif
}
__device__ __forceinline__ void stage_wait() {
#if USE_ASYNC_STAGING
  asm volatile("s_wait_asynccnt 0x0" ::: "memory");
#endif
}

// A fragment (16x32, 16-bit): lane m = lane&15, sel = lane>>4,
// element e -> K = (e&7) + (e>>3)*16 + sel*8   (two 16B runs)
__device__ __forceinline__ v16bf frag_a(const bf16_t* lds, int rowBase, int kk, int lane) {
  const int m = lane & 15, sel = lane >> 4;
  const bf16_t* p = lds + (rowBase + m) * PA + kk + sel * 8;
  v8bf x0 = *(const v8bf*)(p);
  v8bf x1 = *(const v8bf*)(p + 16);
  v16bf r;
#pragma unroll
  for (int e = 0; e < 8; ++e) { r[e] = x0[e]; r[e + 8] = x1[e]; }
  return r;
}

// B fragment (32x16, 16-bit): lane n = lane&15, sel = lane>>4,
// element e -> K = (e&7) + (e>>3)*16 + sel*8.  LDS holds B^T row-major (two 16B runs).
__device__ __forceinline__ v16bf frag_b(const bf16_t* lds, int colBase, int kk, int lane) {
  const int n = lane & 15, sel = lane >> 4;
  const bf16_t* p = lds + (colBase + n) * PB + kk + sel * 8;
  v8bf x0 = *(const v8bf*)(p);
  v8bf x1 = *(const v8bf*)(p + 16);
  v16bf r;
#pragma unroll
  for (int e = 0; e < 8; ++e) { r[e] = x0[e]; r[e + 8] = x1[e]; }
  return r;
}

// Sixteen 16x16x32 WMMAs (two k=32 halves) of one k-step for a wave's 32x64 sub-tile.
__device__ __forceinline__ void wmma_step(const bf16_t* Al, const bf16_t* Bl,
                                          int rowW, int colW, int lane,
                                          v8f acc[2][4]) {
#pragma unroll
  for (int kk = 0; kk < BK; kk += 32) {
    v16bf af[2], bfr[4];
#pragma unroll
    for (int i = 0; i < 2; ++i) af[i] = frag_a(Al, rowW + i * 16, kk, lane);
#pragma unroll
    for (int j = 0; j < 4; ++j) bfr[j] = frag_b(Bl, colW + j * 16, kk, lane);
#pragma unroll
    for (int i = 0; i < 2; ++i) {
#pragma unroll
      for (int j = 0; j < 4; ++j) {
        acc[i][j] = __builtin_amdgcn_wmma_f32_16x16x32_bf16(
            false, af[i], false, bfr[j], (short)0, acc[i][j], false, false);
      }
    }
  }
}

// ---------------------------------------------------------------------------
// Generic GEMM  C[M x 512] = A_bf[M x 512] @ B  (B passed pre-transposed,
// BT[n][k] row-major).  Epilogue modes:
//  0: out fp32 (+bias)                              (input FC)
//  1: out bf16 row-major (+bias) + sum(q^2)         (Q projection)
//  2: out bf16 TRANSPOSED [DIM x MPAD] (+bias) + sum(k^2) + colsum (K proj)
//  3: out bf16 TRANSPOSED [DIM x MPAD] (+bias)      (V projection)
//  4: attn combine: o=(acc*s + Nf*V^T)/(normr+eps); h=(o+h)*0.5 in place
// ---------------------------------------------------------------------------
template <int MODE>
__global__ __launch_bounds__(256) void gemm_bf16_kernel(
    const bf16_t* __restrict__ A, int lda,
    const bf16_t* __restrict__ BT, int ldbt,
    const float* __restrict__ bias,
    float* __restrict__ outF, bf16_t* __restrict__ outBf, int ldc,
    int M, int Kd,
    float* __restrict__ sumsq, float* __restrict__ colsumG,
    const bf16_t* __restrict__ vT, const float* __restrict__ normr,
    const float* __restrict__ scal, float* __restrict__ hio, float Nf) {
  __shared__ bf16_t Alds[2][BM * PA];
  __shared__ bf16_t Blds[2][BN * PB];
  __shared__ float  colsumLds[BN];
  __shared__ float  sqLds;

  const int tid  = threadIdx.x;
  const int lane = tid & 31;
  const int wave = tid >> 5;
  const int rowW = (wave & 3) * 32;   // wave origin in tile (m)
  const int colW = (wave >> 2) * 64;  // wave origin in tile (n)
  const int blockM = blockIdx.y * BM;
  const int blockN = blockIdx.x * BN;

  if (MODE == 2) { if (tid < BN) colsumLds[tid] = 0.0f; }
  if (tid == 0) sqLds = 0.0f;

  v8f acc[2][4];
#pragma unroll
  for (int i = 0; i < 2; ++i) {
#pragma unroll
    for (int j = 0; j < 4; ++j) {
#pragma unroll
      for (int e = 0; e < 8; ++e) acc[i][j][e] = 0.0f;
    }
  }

  // Stage one BK-deep tile pair (A guarded with LDS zeros on the M tail).
  auto stage_tiles = [&](int buf, int kb) {
#pragma unroll
    for (int it = 0; it < 4; ++it) {
      const int id = tid + it * 256;
      const int r = id >> 3;          // row 0..127
      const int c8 = (id & 7) * 8;    // k chunk
      const int grow = blockM + r;
      if (grow < M) stage16(A + (size_t)grow * lda + kb + c8, &Alds[buf][r * PA + c8]);
      else          *(v8bf*)(&Alds[buf][r * PA + c8]) = zero8();
      stage16(BT + (size_t)(blockN + r) * ldbt + kb + c8, &Blds[buf][r * PB + c8]);
    }
  };

  // Pipelined main loop: DMA for k-step i+1 overlaps WMMA of k-step i.
  stage_tiles(0, 0);
  stage_wait();
  __syncthreads();
  int buf = 0;
  for (int kb = 0; kb < Kd; kb += BK) {
    if (kb + BK < Kd) stage_tiles(buf ^ 1, kb + BK);
    wmma_step(&Alds[buf][0], &Blds[buf][0], rowW, colW, lane, acc);
    stage_wait();
    __syncthreads();
    buf ^= 1;
  }

  // ---- Epilogue ----
  const float s = (MODE == 4) ? scal[0] : 0.0f;
  float sqpart = 0.0f;
#pragma unroll
  for (int i = 0; i < 2; ++i) {
#pragma unroll
    for (int j = 0; j < 4; ++j) {
      const int gc  = blockN + colW + j * 16 + (lane & 15);
      const int gr0 = blockM + rowW + i * 16 + (lane >> 4) * 8;
      const float bv = (MODE <= 3) ? bias[gc] : 0.0f;
      if (MODE == 2 || MODE == 3) {
        // Transposed store: a lane owns 8 consecutive rows of one column ->
        // one contiguous 16B store into [DIM x MPAD]; pad rows get zeros.
        float cpart = 0.0f;
        v8bf pack;
#pragma unroll
        for (int r = 0; r < 8; ++r) {
          const int gr = gr0 + r;
          float v = acc[i][j][r] + bv;
          if (gr >= M) v = 0.0f;
          pack[r] = (bf16_t)v;
          if (MODE == 2 && gr < M) { sqpart += v * v; cpart += v; }
        }
        *(v8bf*)(outBf + (size_t)gc * MPAD + gr0) = pack;
        if (MODE == 2) atomicAdd(&colsumLds[colW + j * 16 + (lane & 15)], cpart);
      } else {
#pragma unroll
        for (int r = 0; r < 8; ++r) {
          const int gr = gr0 + r;
          if (gr < M) {
            const float v = acc[i][j][r] + bv;
            const size_t idx = (size_t)gr * ldc + gc;
            if (MODE == 0) {
              outF[idx] = v;
            } else if (MODE == 1) {
              outBf[idx] = (bf16_t)v;
              sqpart += v * v;
            } else if (MODE == 4) {
              const float vv = (float)vT[(size_t)gc * MPAD + gr];
              const float o = (v * s + Nf * vv) / (normr[gr] + 1e-6f);
              hio[idx] = (o + hio[idx]) * 0.5f;
            }
          }
        }
      }
    }
  }
  if (MODE == 1 || MODE == 2) {
    atomicAdd(&sqLds, sqpart);
    __syncthreads();
    if (tid == 0) atomicAdd(&sumsq[MODE == 1 ? 0 : 1], sqLds);
    if (MODE == 2 && tid < BN) atomicAdd(&colsumG[blockN + tid], colsumLds[tid]);
  }
}

// ---------------------------------------------------------------------------
// kvs[512x512] += K^T @ V over an N-slab (split-K, fp32 global atomics).
// Operands are the already-transposed K^T / V^T [DIM x MPAD] buffers, so both
// stage row-major with no guards (pad rows are zero).  Same double-buffered
// async pipeline.
// ---------------------------------------------------------------------------
__global__ __launch_bounds__(256) void ktv_kernel(
    const bf16_t* __restrict__ KT, const bf16_t* __restrict__ VT,
    float* __restrict__ kvs, int slab) {
  __shared__ bf16_t Alds[2][BM * PA];
  __shared__ bf16_t Blds[2][BN * PB];

  const int tid  = threadIdx.x;
  const int lane = tid & 31;
  const int wave = tid >> 5;
  const int rowW = (wave & 3) * 32;
  const int colW = (wave >> 2) * 64;
  const int blockM = blockIdx.y * BM;  // K-feature rows of kvs
  const int blockN = blockIdx.x * BN;  // V-feature cols of kvs
  const int n0 = blockIdx.z * slab;
  int n1 = n0 + slab;
  if (n1 > MPAD) n1 = MPAD;

  v8f acc[2][4];
#pragma unroll
  for (int i = 0; i < 2; ++i) {
#pragma unroll
    for (int j = 0; j < 4; ++j) {
#pragma unroll
      for (int e = 0; e < 8; ++e) acc[i][j][e] = 0.0f;
    }
  }

  auto stage_tiles = [&](int buf, int kb) {
#pragma unroll
    for (int it = 0; it < 4; ++it) {
      const int id = tid + it * 256;
      const int r = id >> 3;          // feature row 0..127
      const int c8 = (id & 7) * 8;    // n chunk
      stage16(KT + (size_t)(blockM + r) * MPAD + kb + c8, &Alds[buf][r * PA + c8]);
      stage16(VT + (size_t)(blockN + r) * MPAD + kb + c8, &Blds[buf][r * PB + c8]);
    }
  };

  stage_tiles(0, n0);
  stage_wait();
  __syncthreads();
  int buf = 0;
  for (int kb = n0; kb < n1; kb += BK) {
    if (kb + BK < n1) stage_tiles(buf ^ 1, kb + BK);
    wmma_step(&Alds[buf][0], &Blds[buf][0], rowW, colW, lane, acc);
    stage_wait();
    __syncthreads();
    buf ^= 1;
  }

#pragma unroll
  for (int i = 0; i < 2; ++i) {
#pragma unroll
    for (int j = 0; j < 4; ++j) {
      const int gc  = blockN + colW + j * 16 + (lane & 15);
      const int gr0 = blockM + rowW + i * 16 + (lane >> 4) * 8;
#pragma unroll
      for (int r = 0; r < 8; ++r)
        atomicAdd(&kvs[(size_t)(gr0 + r) * DIM + gc], acc[i][j][r]);
    }
  }
}

// ---------------------------------------------------------------------------
// Small helper kernels
// ---------------------------------------------------------------------------
__global__ void zero_kernel(float* __restrict__ p, int n) {
  for (int i = blockIdx.x * blockDim.x + threadIdx.x; i < n; i += gridDim.x * blockDim.x)
    p[i] = 0.0f;
}

__global__ void cvt_f32_bf16_kernel(const float* __restrict__ src, bf16_t* __restrict__ dst, int n) {
  for (int i = blockIdx.x * blockDim.x + threadIdx.x; i < n; i += gridDim.x * blockDim.x)
    dst[i] = (bf16_t)src[i];
}

// 512x512 fp32 -> bf16 transpose: dst[n*512+k] = src[k*512+n].  32x32 tiles.
__global__ __launch_bounds__(256) void cvt_transpose_kernel(
    const float* __restrict__ src, bf16_t* __restrict__ dst) {
  __shared__ float tile[32][33];
  const int bx = blockIdx.x * 32;   // n base
  const int by = blockIdx.y * 32;   // k base
  const int tx = threadIdx.x;       // 0..31
  const int ty = threadIdx.y;       // 0..7
#pragma unroll
  for (int i = 0; i < 32; i += 8)
    tile[ty + i][tx] = src[(size_t)(by + ty + i) * DIM + bx + tx];
  __syncthreads();
#pragma unroll
  for (int i = 0; i < 32; i += 8)
    dst[(size_t)(bx + ty + i) * DIM + by + tx] = (bf16_t)tile[tx][ty + i];
}

__global__ void finalize_kernel(const float* __restrict__ sumsq, float* __restrict__ scal) {
  if (threadIdx.x == 0 && blockIdx.x == 0) {
    const float nq = sqrtf(sumsq[0]) + 1e-6f;
    const float nk = sqrtf(sumsq[1]) + 1e-6f;
    scal[0] = 1.0f / (nq * nk);
  }
}

// normr[row] = dot(Q_raw[row,:], colsum(K_raw)) * scal + Nf   (one wave / row)
__global__ __launch_bounds__(256) void normalizer_kernel(
    const bf16_t* __restrict__ qbf, const float* __restrict__ colsum,
    const float* __restrict__ scal, float* __restrict__ normr, float Nf) {
  const int row = blockIdx.x * 8 + (threadIdx.x >> 5);
  const int lane = threadIdx.x & 31;
  if (row >= N_ROWS) return;
  const bf16_t* q = qbf + (size_t)row * DIM;
  float acc = 0.0f;
#pragma unroll 4
  for (int c = lane; c < DIM; c += 32) acc += (float)q[c] * colsum[c];
#pragma unroll
  for (int o = 16; o > 0; o >>= 1) acc += __shfl_down(acc, o, 32);
  if (lane == 0) normr[row] = acc * scal[0] + Nf;
}

// Row-wise LayerNorm + ReLU, in place on h (fp32); emits bf16 copy and
// optionally the final output buffer.  One 128-thread block per row.
__global__ __launch_bounds__(128) void ln_relu_kernel(
    float* __restrict__ h, bf16_t* __restrict__ hbf, float* __restrict__ out2,
    const float* __restrict__ g, const float* __restrict__ b) {
  __shared__ float s1[128];
  __shared__ float s2[128];
  const int row = blockIdx.x;
  const int tid = threadIdx.x;
  float* hr = h + (size_t)row * DIM;
  float x[4];
  float sum = 0.0f, sq = 0.0f;
#pragma unroll
  for (int i = 0; i < 4; ++i) {
    x[i] = hr[tid + i * 128];
    sum += x[i];
    sq += x[i] * x[i];
  }
  s1[tid] = sum;
  s2[tid] = sq;
  __syncthreads();
  for (int o = 64; o > 0; o >>= 1) {
    if (tid < o) { s1[tid] += s1[tid + o]; s2[tid] += s2[tid + o]; }
    __syncthreads();
  }
  const float mean = s1[0] * (1.0f / DIM);
  const float var  = s2[0] * (1.0f / DIM) - mean * mean;
  const float inv  = rsqrtf(var + 1e-5f);
#pragma unroll
  for (int i = 0; i < 4; ++i) {
    const int c = tid + i * 128;
    float y = (x[i] - mean) * inv * g[c] + b[c];
    y = fmaxf(y, 0.0f);
    hr[c] = y;
    hbf[(size_t)row * DIM + c] = (bf16_t)y;
    if (out2) out2[(size_t)row * DIM + c] = y;
  }
}

// ---------------------------------------------------------------------------
// Host orchestration
// ---------------------------------------------------------------------------
extern "C" void kernel_launch(void* const* d_in, const int* in_sizes, int n_in,
                              void* d_out, int out_size, void* d_ws, size_t ws_size,
                              hipStream_t stream) {
  (void)in_sizes; (void)n_in; (void)out_size; (void)ws_size;
  const float* x      = (const float*)d_in[0];
  const float* infc_w = (const float*)d_in[1];
  const float* infc_b = (const float*)d_in[2];
  const float* inln_g = (const float*)d_in[3];
  const float* inln_b = (const float*)d_in[4];
  const float* Wq     = (const float*)d_in[5];   // [2,512,512]
  const float* bq     = (const float*)d_in[6];   // [2,512]
  const float* Wk     = (const float*)d_in[7];
  const float* bk     = (const float*)d_in[8];
  const float* Wv     = (const float*)d_in[9];
  const float* bv     = (const float*)d_in[10];
  const float* ln_g   = (const float*)d_in[11];  // [2,512]
  const float* ln_b   = (const float*)d_in[12];
  float* out = (float*)d_out;

  const size_t nd = (size_t)N_ROWS * DIM;
  const size_t md = (size_t)MPAD * DIM;
  const size_t dd = (size_t)DIM * DIM;
  char* p = (char*)d_ws;
  auto carve = [&](size_t bytes) -> char* {
    char* r = p;
    p += (bytes + 255) & ~(size_t)255;
    return r;
  };
  float*  hF    = (float*)carve(nd * 4);      // residual / pre-LN stream
  bf16_t* hBf   = (bf16_t*)carve(nd * 2);     // GEMM input copy
  bf16_t* qBf   = (bf16_t*)carve(nd * 2);     // also x_bf before layer 0
  bf16_t* kT    = (bf16_t*)carve(md * 2);     // K^T [DIM x MPAD]
  bf16_t* vT    = (bf16_t*)carve(md * 2);     // V^T [DIM x MPAD]
  bf16_t* wT    = (bf16_t*)carve(7 * dd * 2); // transposed weights
  float*  stats = (float*)carve((dd + DIM + 4) * 4);
  bf16_t* kvsT  = (bf16_t*)carve(dd * 2);     // kvs^T bf16
  float*  normr = (float*)carve((size_t)N_ROWS * 4);

  float* kvsF   = stats;
  float* colsum = stats + dd;
  float* sumsq  = colsum + DIM;
  float* scal   = sumsq + 2;

  bf16_t* wInfcT = wT;
  bf16_t* wQT    = wT + dd;       // layers contiguous
  bf16_t* wKT    = wT + 3 * dd;
  bf16_t* wVT    = wT + 5 * dd;

  const dim3 gemmGrid(DIM / BN, (N_ROWS + BM - 1) / BM);  // 4 x 782
  const dim3 trGrid(DIM / 32, DIM / 32);
  const dim3 trBlock(32, 8);
  const float Nf = (float)N_ROWS;

  // --- weight conversion (transposed) + input conversion ---
  cvt_transpose_kernel<<<trGrid, trBlock, 0, stream>>>(infc_w, wInfcT);
  for (int l = 0; l < 2; ++l) {
    cvt_transpose_kernel<<<trGrid, trBlock, 0, stream>>>(Wq + (size_t)l * dd, wQT + (size_t)l * dd);
    cvt_transpose_kernel<<<trGrid, trBlock, 0, stream>>>(Wk + (size_t)l * dd, wKT + (size_t)l * dd);
    cvt_transpose_kernel<<<trGrid, trBlock, 0, stream>>>(Wv + (size_t)l * dd, wVT + (size_t)l * dd);
  }
  cvt_f32_bf16_kernel<<<4096, 256, 0, stream>>>(x, qBf, (int)nd);

  // --- input FC + LN/ReLU ---
  gemm_bf16_kernel<0><<<gemmGrid, 256, 0, stream>>>(
      qBf, DIM, wInfcT, DIM, infc_b, hF, nullptr, DIM, N_ROWS, DIM,
      nullptr, nullptr, nullptr, nullptr, nullptr, nullptr, Nf);
  ln_relu_kernel<<<N_ROWS, 128, 0, stream>>>(hF, hBf, nullptr, inln_g, inln_b);

  for (int l = 0; l < 2; ++l) {
    bf16_t* wq = wQT + (size_t)l * dd;
    bf16_t* wk = wKT + (size_t)l * dd;
    bf16_t* wv = wVT + (size_t)l * dd;

    zero_kernel<<<512, 256, 0, stream>>>(stats, (int)(dd + DIM + 4));

    gemm_bf16_kernel<1><<<gemmGrid, 256, 0, stream>>>(
        hBf, DIM, wq, DIM, bq + l * DIM, nullptr, qBf, DIM, N_ROWS, DIM,
        sumsq, nullptr, nullptr, nullptr, nullptr, nullptr, Nf);
    gemm_bf16_kernel<2><<<gemmGrid, 256, 0, stream>>>(
        hBf, DIM, wk, DIM, bk + l * DIM, nullptr, kT, DIM, N_ROWS, DIM,
        sumsq, colsum, nullptr, nullptr, nullptr, nullptr, Nf);
    gemm_bf16_kernel<3><<<gemmGrid, 256, 0, stream>>>(
        hBf, DIM, wv, DIM, bv + l * DIM, nullptr, vT, DIM, N_ROWS, DIM,
        nullptr, nullptr, nullptr, nullptr, nullptr, nullptr, Nf);

    const int slab = 512;
    const int nz = (MPAD + slab - 1) / slab;  // 196
    ktv_kernel<<<dim3(DIM / BN, DIM / BM, nz), 256, 0, stream>>>(kT, vT, kvsF, slab);

    finalize_kernel<<<1, 64, 0, stream>>>(sumsq, scal);
    cvt_transpose_kernel<<<trGrid, trBlock, 0, stream>>>(kvsF, kvsT);
    normalizer_kernel<<<(N_ROWS + 7) / 8, 256, 0, stream>>>(qBf, colsum, scal, normr, Nf);

    gemm_bf16_kernel<4><<<gemmGrid, 256, 0, stream>>>(
        qBf, DIM, kvsT, DIM, nullptr, nullptr, nullptr, DIM, N_ROWS, DIM,
        nullptr, nullptr, vT, normr, scal, hF, Nf);

    ln_relu_kernel<<<N_ROWS, 128, 0, stream>>>(
        hF, hBf, (l == 1) ? out : nullptr, ln_g + l * DIM, ln_b + l * DIM);
  }
}
